// GrowableAttention_84284438216952
// MI455X (gfx1250) — matrix operand order repaired and checked by reading, not compiled, not run on hardware
//
#include <hip/hip_runtime.h>

// ---------------------------------------------------------------------------
// Types for CDNA5 WMMA (wave32, 16x16x32 bf16 -> f32)
// ---------------------------------------------------------------------------
typedef __attribute__((ext_vector_type(8)))  __bf16 v8bf;
typedef __attribute__((ext_vector_type(16))) __bf16 v16bf;
typedef __attribute__((ext_vector_type(8)))  float  v8f;

static constexpr int Sx   = 2048;
static constexpr int NH   = 16;
static constexpr int NKV  = 4;
static constexpr int HD   = 128;

static __device__ __forceinline__ v8f vzero() {
  v8f z = {0.f, 0.f, 0.f, 0.f, 0.f, 0.f, 0.f, 0.f};
  return z;
}

// f32 -> bf16 (round to nearest even), raw u16
static __device__ __forceinline__ unsigned f2bf(float f) {
  unsigned u = __float_as_uint(f);
  return (u + 0x7FFFu + ((u >> 16) & 1u)) >> 16;
}
static __device__ __forceinline__ unsigned pack2bf(float lo, float hi) {
  return (f2bf(lo) & 0xFFFFu) | (f2bf(hi) << 16);
}

// Build a 16-element bf16 fragment from two contiguous 8-element (16B) chunks
static __device__ __forceinline__ v16bf frag16(const unsigned short* p0,
                                               const unsigned short* p1) {
  v8bf lo = *reinterpret_cast<const v8bf*>(p0);
  v8bf hi = *reinterpret_cast<const v8bf*>(p1);
  v16bf r;
#pragma unroll
  for (int i = 0; i < 8; ++i) { r[i] = lo[i]; r[i + 8] = hi[i]; }
  return r;
}

static __device__ __forceinline__ v8f wmma_bf16(v16bf a, v16bf b, v8f c) {
  return __builtin_amdgcn_wmma_f32_16x16x32_bf16(
      /*neg_a=*/false, a, /*neg_b=*/false, b,
      /*c_mod=*/(short)0, c, /*reuse_a=*/false, /*reuse_b=*/false);
}

// Async global->LDS copy (16B per lane), tracked by ASYNCcnt.
static __device__ __forceinline__ void async_g2l_b128(void* lds_ptr,
                                                      const void* gptr) {
  unsigned lds_off = (unsigned)(uintptr_t)lds_ptr;  // generic LDS = addr[31:0]
  unsigned long long ga = (unsigned long long)(uintptr_t)gptr;
  asm volatile("global_load_async_to_lds_b128 %0, %1, off"
               :: "v"(lds_off), "v"(ga) : "memory");
}
static __device__ __forceinline__ void wait_async0() {
  asm volatile("s_wait_asynccnt 0" ::: "memory");
}

// Half-wave (16-lane) reductions via ds_swizzle_b32 (group-of-32 xor masks)
#define SWZF(x, imm) \
  __int_as_float(__builtin_amdgcn_ds_swizzle(__float_as_int(x), (imm)))

static __device__ __forceinline__ float rmax16(float x) {
  x = fmaxf(x, SWZF(x, (1 << 10) | 0x1f));
  x = fmaxf(x, SWZF(x, (2 << 10) | 0x1f));
  x = fmaxf(x, SWZF(x, (4 << 10) | 0x1f));
  x = fmaxf(x, SWZF(x, (8 << 10) | 0x1f));
  return x;
}
static __device__ __forceinline__ float rsum16(float x) {
  x += SWZF(x, (1 << 10) | 0x1f);
  x += SWZF(x, (2 << 10) | 0x1f);
  x += SWZF(x, (4 << 10) | 0x1f);
  x += SWZF(x, (8 << 10) | 0x1f);
  return x;
}

// ---------------------------------------------------------------------------
// bf16 x bf16 GEMM: C[M,N] (f32) = A[M,K] * B[N,K]^T, A/B bf16 row-major.
// Block tile 128x128x32, 8 waves, each wave: 4x2 grid of 16x16 WMMA tiles.
// Double-buffered LDS; tiles staged with GLOBAL_LOAD_ASYNC_TO_LDS_B128 so the
// copy of K-step k+1 overlaps the WMMA work of K-step k. One barrier / step.
// ---------------------------------------------------------------------------
__global__ void __launch_bounds__(256)
wmma_gemm_bb(const unsigned short* __restrict__ A,
             const unsigned short* __restrict__ Bw,
             float* __restrict__ C, int M, int N, int K) {
  constexpr int BK = 32, LDT = 40;  // 80B row stride: conflict-free frag rows
  __shared__ __align__(16) unsigned short As[2][128 * LDT];
  __shared__ __align__(16) unsigned short Bs[2][128 * LDT];

  const int tid  = threadIdx.x;
  const int lane = tid & 31;
  const int wave = tid >> 5;
  const int wm = (wave & 1) * 64;   // 2 waves along M
  const int wn = (wave >> 1) * 32;  // 4 waves along N
  const int bm = blockIdx.y * 128;
  const int bn = blockIdx.x * 128;

  v8f acc[4][2];
#pragma unroll
  for (int i = 0; i < 4; ++i)
#pragma unroll
    for (int j = 0; j < 2; ++j) acc[i][j] = vzero();

  // Issue one 128x32 bf16 tile pair as raw 16B async copies (4 per thread).
  auto issue_tile = [&](int buf, int k0) {
#pragma unroll
    for (int i = 0; i < 2; ++i) {
      int e = tid + i * 256;           // 512 chunks of 8 ushorts per matrix
      int r = e >> 2, c8 = (e & 3) * 8;
      async_g2l_b128(&As[buf][r * LDT + c8],
                     &A[(size_t)(bm + r) * K + k0 + c8]);
      async_g2l_b128(&Bs[buf][r * LDT + c8],
                     &Bw[(size_t)(bn + r) * K + k0 + c8]);
    }
  };

  issue_tile(0, 0);

  const int kg0 = (lane & 16) ? 8 : 0;
  const int bh  = (lane & 16) ? 8 : 0;

  for (int k0 = 0, it = 0; k0 < K; k0 += BK, ++it) {
    const int cur = it & 1;
    wait_async0();        // my copies for buf[cur] have landed
    __syncthreads();      // everyone's copies landed; prev reads of buf[cur^1] done
    if (k0 + BK < K) issue_tile(cur ^ 1, k0 + BK);  // overlaps compute below

    v16bf afr[4], bfr[2];
#pragma unroll
    for (int i = 0; i < 4; ++i) {
      const unsigned short* pr = &As[cur][(wm + i * 16 + (lane & 15)) * LDT];
      afr[i] = frag16(pr + kg0, pr + kg0 + 16);
    }
#pragma unroll
    for (int j = 0; j < 2; ++j) {
      const unsigned short* pr = &Bs[cur][(wn + j * 16 + (lane & 15)) * LDT + bh];
      bfr[j] = frag16(pr, pr + 16);
    }
#pragma unroll
    for (int i = 0; i < 4; ++i)
#pragma unroll
      for (int j = 0; j < 2; ++j)
        acc[i][j] = wmma_bf16(afr[i], bfr[j], acc[i][j]);
  }

  // C/D layout: element (m,n): lane = n + 16*(m/8), vgpr = m%8.
  const int half = lane >> 4, nn = lane & 15;
#pragma unroll
  for (int i = 0; i < 4; ++i)
#pragma unroll
    for (int j = 0; j < 2; ++j)
#pragma unroll
      for (int v = 0; v < 8; ++v) {
        int m = bm + wm + i * 16 + v + 8 * half;
        C[(size_t)m * N + bn + wn + j * 16 + nn] = acc[i][j][v];
      }
}

// ---------------------------------------------------------------------------
// Vectorized f32 -> bf16 conversion (float4 -> packed uint2).
// ---------------------------------------------------------------------------
__global__ void cvt_bf16x4(const float4* __restrict__ src,
                           uint2* __restrict__ dst, int n4) {
  int i = blockIdx.x * blockDim.x + threadIdx.x;
  if (i >= n4) return;
  float4 v = src[i];
  uint2 p;
  p.x = pack2bf(v.x, v.y);
  p.y = pack2bf(v.z, v.w);
  dst[i] = p;
}

// ---------------------------------------------------------------------------
// RoPE (half-split rotation) fused with f32 -> bf16 conversion.
// ---------------------------------------------------------------------------
__global__ void rope_convert(const float* __restrict__ src,
                             unsigned short* __restrict__ dst,
                             int nheads, int total_pairs) {
  int idx = blockIdx.x * blockDim.x + threadIdx.x;
  if (idx >= total_pairs) return;
  int d = idx & 63;                 // pair index
  int h = (idx >> 6) % nheads;
  int m = idx / (64 * nheads);      // b*S + s
  int s = m & (Sx - 1);
  size_t base = (size_t)m * (nheads * HD) + (size_t)h * HD + d;
  float x1 = src[base], x2 = src[base + 64];
  float freq = (float)s * __powf(10000.0f, -(float)(2 * d) * (1.0f / 128.0f));
  float sn, cs;
  __sincosf(freq, &sn, &cs);
  dst[base]      = (unsigned short)f2bf(x1 * cs - x2 * sn);
  dst[base + 64] = (unsigned short)f2bf(x2 * cs + x1 * sn);
}

// ---------------------------------------------------------------------------
// Flash attention (no mask), GQA 16q/4kv heads, d=128.
// Block = 128 q rows x one head x one batch; 8 waves, 16 q rows each.
// K tile staged via GLOBAL_LOAD_ASYNC_TO_LDS_B128 (ASYNCcnt path).
// ---------------------------------------------------------------------------
__global__ void __launch_bounds__(256)
flash_attn(const unsigned short* __restrict__ Qb,
           const unsigned short* __restrict__ Kb,
           const unsigned short* __restrict__ Vb,
           unsigned short* __restrict__ Ob) {
  constexpr int KT = 64;
  constexpr int LDK = 136;  // [key][d] stride (ushorts), 16B aligned rows
  constexpr int LDV = 72;   // [d][key] and [qrow][key] stride
  __shared__ __align__(16) unsigned short Ks[KT * LDK];
  __shared__ __align__(16) unsigned short Vt[HD * LDV];
  __shared__ __align__(16) unsigned short Ps[128 * LDV];

  const int tid  = threadIdx.x;
  const int lane = tid & 31;
  const int wave = tid >> 5;
  const int q0   = blockIdx.x * 128;
  const int head = blockIdx.y;
  const int b    = blockIdx.z;
  const int kvh  = head >> 2;
  const float scale = 0.08838834764831845f;  // 1/sqrt(128)

  // Preload this wave's Q A-fragments (16 rows x 128 d) from global.
  v16bf qf[4];
  {
    const size_t rb =
        (size_t)(b * Sx + q0 + wave * 16 + (lane & 15)) * (NH * HD) +
        (size_t)head * HD;
    const int kg0 = (lane & 16) ? 8 : 0;
#pragma unroll
    for (int ks = 0; ks < 4; ++ks) {
      const unsigned short* p = Qb + rb + ks * 32 + kg0;
      qf[ks] = frag16(p, p + 16);
    }
  }

  v8f o[8];
#pragma unroll
  for (int nt = 0; nt < 8; ++nt) o[nt] = vzero();
  float mi[8], li[8];
#pragma unroll
  for (int v = 0; v < 8; ++v) { mi[v] = -3.0e38f; li[v] = 0.f; }

  for (int kb = 0; kb < Sx; kb += KT) {
    __syncthreads();
    // K tile [key][d]: async global->LDS, 16B per lane, 4 chunks per thread.
#pragma unroll
    for (int i = 0; i < 4; ++i) {
      int e = tid + i * 256;          // 1024 chunks of 8 ushorts
      int r = e >> 4, c = (e & 15) * 8;
      async_g2l_b128(
          &Ks[r * LDK + c],
          &Kb[(size_t)(b * Sx + kb + r) * (NKV * HD) + (size_t)kvh * HD + c]);
    }
    // V tile transposed [d][key]: gather 8 keys per chunk, one ds_store_b128.
#pragma unroll
    for (int i = 0; i < 4; ++i) {
      int e = tid + i * 256;          // 1024 chunks
      int d = e & 127, kg = (e >> 7) * 8;
      const unsigned short* src =
          &Vb[(size_t)(b * Sx + kb + kg) * (NKV * HD) + (size_t)kvh * HD + d];
      unsigned t0 = src[0 * 512], t1 = src[1 * 512], t2 = src[2 * 512],
               t3 = src[3 * 512], t4 = src[4 * 512], t5 = src[5 * 512],
               t6 = src[6 * 512], t7 = src[7 * 512];
      uint4 p;
      p.x = t0 | (t1 << 16);
      p.y = t2 | (t3 << 16);
      p.z = t4 | (t5 << 16);
      p.w = t6 | (t7 << 16);
      *reinterpret_cast<uint4*>(&Vt[d * LDV + kg]) = p;
    }
    wait_async0();
    __syncthreads();

    // Scores: S(16x64) = Q(16x128) * K^T. Batch B-frag loads per k-step.
    v8f sacc[4];
#pragma unroll
    for (int nt = 0; nt < 4; ++nt) sacc[nt] = vzero();
    const int bh = (lane & 16) ? 8 : 0;
#pragma unroll
    for (int ks = 0; ks < 4; ++ks) {
      v16bf kfr[4];
#pragma unroll
      for (int nt = 0; nt < 4; ++nt) {
        const unsigned short* pr =
            &Ks[(nt * 16 + (lane & 15)) * LDK + ks * 32 + bh];
        kfr[nt] = frag16(pr, pr + 16);
      }
#pragma unroll
      for (int nt = 0; nt < 4; ++nt)
        sacc[nt] = wmma_bf16(qf[ks], kfr[nt], sacc[nt]);
    }

    // Online softmax: row m = vgpr + 8*(lane/16); reduce across 16 lanes.
    float corr[8];
#pragma unroll
    for (int v = 0; v < 8; ++v) {
      float rm = fmaxf(fmaxf(sacc[0][v], sacc[1][v]),
                       fmaxf(sacc[2][v], sacc[3][v]));
      rm = rmax16(rm) * scale;
      float mn = fmaxf(mi[v], rm);
      corr[v] = __expf(mi[v] - mn);
      mi[v] = mn;
      float rs = 0.f;
#pragma unroll
      for (int nt = 0; nt < 4; ++nt) {
        float p = __expf(sacc[nt][v] * scale - mn);
        sacc[nt][v] = p;
        rs += p;
      }
      li[v] = li[v] * corr[v] + rsum16(rs);
    }
#pragma unroll
    for (int nt = 0; nt < 8; ++nt)
#pragma unroll
      for (int v = 0; v < 8; ++v) o[nt][v] *= corr[v];

    // Stage P (C-layout -> A-layout) through wave-private LDS.
    const int half = lane >> 4, nn = lane & 15;
#pragma unroll
    for (int v = 0; v < 8; ++v) {
      unsigned short* row = &Ps[(wave * 16 + v + 8 * half) * LDV + nn];
#pragma unroll
      for (int nt = 0; nt < 4; ++nt)
        row[nt * 16] = (unsigned short)f2bf(sacc[nt][v]);
    }

    // O(16x128) += P(16x64) * V(64x128): batch Vt frag loads in groups of 4.
    const int kg0 = (lane & 16) ? 8 : 0;
#pragma unroll
    for (int ks = 0; ks < 2; ++ks) {
      const unsigned short* pa =
          &Ps[(wave * 16 + (lane & 15)) * LDV + ks * 32 + kg0];
      v16bf af = frag16(pa, pa + 16);
#pragma unroll
      for (int g = 0; g < 2; ++g) {
        v16bf vfr[4];
#pragma unroll
        for (int j = 0; j < 4; ++j) {
          int nt = g * 4 + j;
          const unsigned short* pb =
              &Vt[(nt * 16 + (lane & 15)) * LDV + ks * 32 + bh];
          vfr[j] = frag16(pb, pb + 16);
        }
#pragma unroll
        for (int j = 0; j < 4; ++j)
          o[g * 4 + j] = wmma_bf16(af, vfr[j], o[g * 4 + j]);
      }
    }
  }

  // Normalize and store bf16 output at [b*S+s][head*128+d] for the out-proj.
  const int half = lane >> 4, nn = lane & 15;
#pragma unroll
  for (int v = 0; v < 8; ++v) {
    float inv = 1.0f / li[v];
    size_t m = (size_t)(b * Sx + q0 + wave * 16 + v + 8 * half);
#pragma unroll
    for (int nt = 0; nt < 8; ++nt)
      Ob[m * (NH * HD) + head * HD + nt * 16 + nn] =
          (unsigned short)f2bf(o[nt][v] * inv);
  }
}

// ---------------------------------------------------------------------------
// Host-side orchestration (graph-capture safe: launches only).
// ---------------------------------------------------------------------------
extern "C" void kernel_launch(void* const* d_in, const int* in_sizes, int n_in,
                              void* d_out, int out_size, void* d_ws, size_t ws_size,
                              hipStream_t stream) {
  (void)in_sizes; (void)n_in; (void)out_size; (void)ws_size;
  const float* hid = (const float*)d_in[0];
  const float* Wq  = (const float*)d_in[1];
  const float* Wk  = (const float*)d_in[2];
  const float* Wv  = (const float*)d_in[3];
  const float* Wo  = (const float*)d_in[4];

  char* ws = (char*)d_ws;
  float*          qf  = (float*)(ws + 0);                  // 4096x2048 f32
  float*          kf  = (float*)(ws + 33554432);           // 4096x512  f32
  float*          vf  = (float*)(ws + 41943040);           // 4096x512  f32
  unsigned short* qb  = (unsigned short*)(ws + 50331648);  // 4096x2048 bf16
  unsigned short* kb  = (unsigned short*)(ws + 67108864);  // 4096x512  bf16
  unsigned short* vb  = (unsigned short*)(ws + 71303168);  // 4096x512  bf16
  unsigned short* ab  = (unsigned short*)(ws + 75497472);  // 4096x2048 bf16
  unsigned short* hb  = (unsigned short*)(ws + 92274688);  // 4096x2048 bf16
  unsigned short* wqb = (unsigned short*)(ws + 109051904); // 2048x2048 bf16
  unsigned short* wkb = (unsigned short*)(ws + 117440512); // 512x2048  bf16
  unsigned short* wvb = (unsigned short*)(ws + 119537664); // 512x2048  bf16
  unsigned short* wob = (unsigned short*)(ws + 121634816); // 2048x2048 bf16

  dim3 blk(256);
  // One-time bf16 conversion of activations + weights (removes per-tile cvt
  // from the GEMM hot loops and halves re-read traffic).
  cvt_bf16x4<<<dim3(8192), blk, 0, stream>>>((const float4*)hid, (uint2*)hb,  2097152);
  cvt_bf16x4<<<dim3(4096), blk, 0, stream>>>((const float4*)Wq,  (uint2*)wqb, 1048576);
  cvt_bf16x4<<<dim3(1024), blk, 0, stream>>>((const float4*)Wk,  (uint2*)wkb, 262144);
  cvt_bf16x4<<<dim3(1024), blk, 0, stream>>>((const float4*)Wv,  (uint2*)wvb, 262144);
  cvt_bf16x4<<<dim3(4096), blk, 0, stream>>>((const float4*)Wo,  (uint2*)wob, 1048576);
  // Projections (C = hidden @ W^T), async double-buffered bf16 GEMMs.
  wmma_gemm_bb<<<dim3(16, 32), blk, 0, stream>>>(hb, wqb, qf, 4096, 2048, 2048);
  wmma_gemm_bb<<<dim3(4, 32),  blk, 0, stream>>>(hb, wkb, kf, 4096, 512, 2048);
  wmma_gemm_bb<<<dim3(4, 32),  blk, 0, stream>>>(hb, wvb, vf, 4096, 512, 2048);
  // RoPE + bf16 conversion
  rope_convert<<<dim3(4194304 / 256), blk, 0, stream>>>(qf, qb, NH,  4194304);
  rope_convert<<<dim3(1048576 / 256), blk, 0, stream>>>(kf, kb, NKV, 1048576);
  cvt_bf16x4<<<dim3(2048), blk, 0, stream>>>((const float4*)vf, (uint2*)vb, 524288);
  // Attention
  flash_attn<<<dim3(16, 16, 2), blk, 0, stream>>>(qb, kb, vb, ab);
  // Output projection -> d_out (f32)
  wmma_gemm_bb<<<dim3(16, 32), blk, 0, stream>>>(ab, wob, (float*)d_out, 4096, 2048, 2048);
}
